// Decoder_75720273428908
// MI455X (gfx1250) — matrix operand order repaired and checked by reading, not compiled, not run on hardware
//
#include <hip/hip_runtime.h>
#include <hip/hip_bf16.h>

// ---------------------------------------------------------------------------
// CDNA5 (gfx1250) implementation: wave32, WMMA bf16 16x16x32, fp32 accumulate.
// Fragment data is 16B-aligned & contiguous in LDS -> ds_load_b128 gathers.
// Interior tiles take a guard-free staging fast path (b128 loads/stores),
// bf16 conversion uses the hardware cvt via (__bf16) casts.
// ---------------------------------------------------------------------------

typedef __attribute__((ext_vector_type(16))) __bf16 v16bf;
typedef __attribute__((ext_vector_type(8)))  __bf16 v8bf;
typedef __attribute__((ext_vector_type(8)))  float  v8f;

#define Bsz 16
#define Tt  512
#define Ss  1024
#define Aa  64
#define Hh  256
#define NHh 8
#define Dd  32
#define PFf 512

__device__ __forceinline__ unsigned pack2bf(float x, float y) {
    union { __bf16 h[2]; unsigned u; } p;
    p.h[0] = (__bf16)x; p.h[1] = (__bf16)y;   // hw v_cvt (RNE)
    return p.u;
}

// ---------------------------------------------------------------------------
// Generic batched GEMM: C = act(A[M,K] x B[K,N] + bias)
//   A fp32 (lda), B fp32 (ldb, stored [N,K] when transB), C fp32 (ldc).
//   Batch offset = zb*OffB + zh*OffH with z = blockIdx.z, zb=z/nh, zh=z%nh.
// Tile: BM=64, BN=64, BK=32. 128 threads = 4 wave32; each wave owns a 16-row
// strip and 4 column fragments -> 4 x v_wmma_f32_16x16x32_bf16 per K-step.
// LDS: As[m][k] pitch 40 (80B, 16B multiple); Bs stored transposed [n][k].
// ---------------------------------------------------------------------------
__global__ __launch_bounds__(128)
void k_gemm(const float* __restrict__ A, const float* __restrict__ Bm,
            const float* __restrict__ bias, float* __restrict__ C,
            int M, int N, int K, int lda, int ldb, int ldc,
            long long aOffB, long long aOffH,
            long long bOffB, long long bOffH,
            long long cOffB, long long cOffH,
            int nh, int transB, int relu)
{
    __shared__ __align__(16) __bf16 As[64][40];  // [m][k], 80B pitch
    __shared__ __align__(16) __bf16 Bs[64][40];  // [n][k], 80B pitch

    const int tid  = threadIdx.x;
    const int wave = tid >> 5;
    const int lane = tid & 31;
    const int tileM = blockIdx.y * 64;
    const int tileN = blockIdx.x * 64;
    const int zb = (int)blockIdx.z / nh;
    const int zh = (int)blockIdx.z % nh;

    const float* Ap = A  + (long long)zb * aOffB + (long long)zh * aOffH;
    const float* Bp = Bm + (long long)zb * bOffB + (long long)zh * bOffH;
    float*       Cp = C  + (long long)zb * cOffB + (long long)zh * cOffH;

    // interior tile: no guards needed anywhere in staging (K always %32==0)
    const bool full = (tileM + 64 <= M) && (tileN + 64 <= N) && ((K & 31) == 0);

    v8f acc[4];
#pragma unroll
    for (int f = 0; f < 4; ++f)
#pragma unroll
        for (int r = 0; r < 8; ++r) acc[f][r] = 0.0f;

    for (int k0 = 0; k0 < K; k0 += 32) {
        if (full) {
            // ---- fast path: thread owns one 16-wide k-segment of a row ----
            {
                int r  = tid >> 1;            // 0..63
                int cs = (tid & 1) << 4;      // 0 or 16
                const float4* s4 = reinterpret_cast<const float4*>(
                    &Ap[(size_t)(tileM + r) * lda + k0 + cs]);
                union { __bf16 h[16]; v8bf q[2]; } t;
#pragma unroll
                for (int j = 0; j < 4; ++j) {
                    float4 v = s4[j];
                    t.h[4 * j + 0] = (__bf16)v.x;
                    t.h[4 * j + 1] = (__bf16)v.y;
                    t.h[4 * j + 2] = (__bf16)v.z;
                    t.h[4 * j + 3] = (__bf16)v.w;
                }
                *reinterpret_cast<v8bf*>(&As[r][cs])     = t.q[0];
                *reinterpret_cast<v8bf*>(&As[r][cs + 8]) = t.q[1];
            }
            if (transB) {
                int n  = tid >> 1;
                int cs = (tid & 1) << 4;
                const float4* s4 = reinterpret_cast<const float4*>(
                    &Bp[(size_t)(tileN + n) * ldb + k0 + cs]);
                union { __bf16 h[16]; v8bf q[2]; } t;
#pragma unroll
                for (int j = 0; j < 4; ++j) {
                    float4 v = s4[j];
                    t.h[4 * j + 0] = (__bf16)v.x;
                    t.h[4 * j + 1] = (__bf16)v.y;
                    t.h[4 * j + 2] = (__bf16)v.z;
                    t.h[4 * j + 3] = (__bf16)v.w;
                }
                *reinterpret_cast<v8bf*>(&Bs[n][cs])     = t.q[0];
                *reinterpret_cast<v8bf*>(&Bs[n][cs + 8]) = t.q[1];
            } else {
                // B stored [K,N]: coalesced reads across n, transpose scatter
#pragma unroll
                for (int i = 0; i < 16; ++i) {
                    int idx = tid + i * 128;
                    int kk = idx >> 6, n = idx & 63;
                    Bs[n][kk] =
                        (__bf16)Bp[(size_t)(k0 + kk) * ldb + (tileN + n)];
                }
            }
        } else {
            // ---- guarded path (edge tiles: the M=16 fc1 and N=32 ctx GEMMs) ----
#pragma unroll
            for (int i = 0; i < 8; ++i) {
                int p  = tid + i * 128;
                int r  = p >> 4;
                int c2 = (p & 15) << 1;
                int gr = tileM + r, gc = k0 + c2;
                unsigned u = 0;
                if (gr < M && gc < K) {
                    const float2 v = *reinterpret_cast<const float2*>(
                        &Ap[(size_t)gr * lda + gc]);
                    u = pack2bf(v.x, v.y);
                }
                *reinterpret_cast<unsigned*>(&As[r][c2]) = u;
            }
            if (transB) {
#pragma unroll
                for (int i = 0; i < 8; ++i) {
                    int p  = tid + i * 128;
                    int n  = p >> 4;
                    int k2 = (p & 15) << 1;
                    int gn = tileN + n, gk = k0 + k2;
                    unsigned u = 0;
                    if (gn < N && gk < K) {
                        const float2 v = *reinterpret_cast<const float2*>(
                            &Bp[(size_t)gn * ldb + gk]);
                        u = pack2bf(v.x, v.y);
                    }
                    *reinterpret_cast<unsigned*>(&Bs[n][k2]) = u;
                }
            } else {
#pragma unroll
                for (int i = 0; i < 16; ++i) {
                    int idx = tid + i * 128;
                    int kk = idx >> 6, n = idx & 63;
                    int gk = k0 + kk, gn = tileN + n;
                    float v = (gk < K && gn < N)
                                  ? Bp[(size_t)gk * ldb + gn] : 0.0f;
                    Bs[n][kk] = (__bf16)v;
                }
            }
        }
        // prefetch next A tile row for this thread (global_prefetch_b8)
        if (k0 + 32 < K) {
            int gr = tileM + (tid >> 5);
            if (gr < M) __builtin_prefetch(&Ap[(size_t)gr * lda + k0 + 32], 0, 0);
        }
        __syncthreads();

        // ---- A fragment: lanes 0-15: K 0-7 & 16-23; lanes 16-31: K 8-15 &
        //      24-31; row m = wave*16 + lane%16.  2x ds_load_b128 ----
        union { v8bf h[2]; v16bf v; } af;
        {
            int mr = (wave << 4) + (lane & 15);
            int kb = (lane >> 4) << 3;       // 0 or 8  (16B-aligned offset)
            af.h[0] = *reinterpret_cast<const v8bf*>(&As[mr][kb]);
            af.h[1] = *reinterpret_cast<const v8bf*>(&As[mr][kb + 16]);
        }
#pragma unroll
        for (int f = 0; f < 4; ++f) {
            // ---- B fragment: lane nc holds K kbb..kbb+7 & kbb+16..kbb+23 in
            //      Bs[n][k].  2x ds_load_b128 ----
            union { v8bf h[2]; v16bf v; } bfv;
            int nc  = (f << 4) + (lane & 15);
            int kbb = (lane >> 4) << 3;      // 0 or 8  (16B-aligned offset)
            bfv.h[0] = *reinterpret_cast<const v8bf*>(&Bs[nc][kbb]);
            bfv.h[1] = *reinterpret_cast<const v8bf*>(&Bs[nc][kbb + 16]);

            acc[f] = __builtin_amdgcn_wmma_f32_16x16x32_bf16(
                false, af.v, false, bfv.v, (short)0, acc[f], false, false);
        }
        __syncthreads();
    }

    // ---- store: C/D layout: VGPR r -> m = r + (lane/16)*8, n = lane%16 ----
#pragma unroll
    for (int f = 0; f < 4; ++f) {
        int gn = tileN + (f << 4) + (lane & 15);
#pragma unroll
        for (int r = 0; r < 8; ++r) {
            int gm = tileM + (wave << 4) + r + ((lane >> 4) << 3);
            if (gm < M && gn < N) {
                float v = acc[f][r];
                if (bias) v += bias[gn];
                if (relu && v < 0.0f) v = 0.0f;
                Cp[(size_t)gm * ldc + gn] = v;
            }
        }
    }
}

// ---------------------------------------------------------------------------
// Row softmax with pre-scale: x <- softmax(x * invs) over `cols` (<=1024).
// ---------------------------------------------------------------------------
__global__ __launch_bounds__(256)
void k_softmax(float* __restrict__ X, int cols, float invs)
{
    const int tid = threadIdx.x;
    float* p = X + (size_t)blockIdx.x * cols;
    __shared__ float red[256];

    float vals[4];
    float m = -3.4e38f;
#pragma unroll
    for (int i = 0; i < 4; ++i) {
        int c = tid + i * 256;
        float v = (c < cols) ? p[c] * invs : -3.4e38f;
        vals[i] = v;
        m = fmaxf(m, v);
    }
    red[tid] = m; __syncthreads();
    for (int s = 128; s > 0; s >>= 1) {
        if (tid < s) red[tid] = fmaxf(red[tid], red[tid + s]);
        __syncthreads();
    }
    m = red[0]; __syncthreads();

    float sum = 0.0f;
#pragma unroll
    for (int i = 0; i < 4; ++i) {
        int c = tid + i * 256;
        if (c < cols) { vals[i] = __expf(vals[i] - m); sum += vals[i]; }
    }
    red[tid] = sum; __syncthreads();
    for (int s = 128; s > 0; s >>= 1) {
        if (tid < s) red[tid] += red[tid + s];
        __syncthreads();
    }
    float inv = 1.0f / red[0]; __syncthreads();
#pragma unroll
    for (int i = 0; i < 4; ++i) {
        int c = tid + i * 256;
        if (c < cols) p[c] = vals[i] * inv;
    }
}

// ---------------------------------------------------------------------------
// x <- LayerNorm(x + r) * g + b  (row length 256, one block per row)
// ---------------------------------------------------------------------------
__global__ __launch_bounds__(256)
void k_add_ln(float* __restrict__ X, const float* __restrict__ R,
              const float* __restrict__ g, const float* __restrict__ b)
{
    const int tid = threadIdx.x;
    const size_t base = (size_t)blockIdx.x * 256 + tid;
    float v = X[base] + R[base];
    __shared__ float red[256];

    red[tid] = v; __syncthreads();
    for (int s = 128; s > 0; s >>= 1) {
        if (tid < s) red[tid] += red[tid + s];
        __syncthreads();
    }
    float mean = red[0] * (1.0f / 256.0f); __syncthreads();
    float d = v - mean;
    red[tid] = d * d; __syncthreads();
    for (int s = 128; s > 0; s >>= 1) {
        if (tid < s) red[tid] += red[tid + s];
        __syncthreads();
    }
    float var = red[0] * (1.0f / 256.0f);
    X[base] = d * rsqrtf(var + 1e-5f) * g[tid] + b[tid];
}

// ---------------------------------------------------------------------------
// row L2 norm (row length 256)
// ---------------------------------------------------------------------------
__global__ __launch_bounds__(256)
void k_rownorm(const float* __restrict__ X, float* __restrict__ out)
{
    const int tid = threadIdx.x;
    float v = X[(size_t)blockIdx.x * 256 + tid];
    __shared__ float red[256];
    red[tid] = v * v; __syncthreads();
    for (int s = 128; s > 0; s >>= 1) {
        if (tid < s) red[tid] += red[tid + s];
        __syncthreads();
    }
    if (tid == 0) out[blockIdx.x] = sqrtf(red[0]);
}

// pooled[b,h] = sum_t w[b,t] * X[b,t,h]
__global__ __launch_bounds__(256)
void k_pool(const float* __restrict__ X, const float* __restrict__ w,
            float* __restrict__ pooled)
{
    const int b = blockIdx.x, h = threadIdx.x;
    float s = 0.0f;
    for (int t = 0; t < Tt; ++t)
        s += w[b * Tt + t] * X[((size_t)(b * Tt + t)) * Hh + h];
    pooled[b * Hh + h] = s;
}

// label[b,c] = fc1out[b,:] @ fc2_w[:,c] + fc2_b[c]
__global__ void k_fc2(const float* __restrict__ fin, const float* __restrict__ w,
                      const float* __restrict__ bias, float* __restrict__ out)
{
    int i = threadIdx.x;
    if (i >= 32) return;
    int b = i >> 1, c = i & 1;
    float s = bias[c];
    for (int k = 0; k < 256; ++k) s += fin[b * 256 + k] * w[k * 2 + c];
    out[b * 2 + c] = s;
}

// ---------------------------------------------------------------------------
// Host-side orchestration
// ---------------------------------------------------------------------------
static inline void gemm(hipStream_t s, const float* A, const float* B,
                        const float* bias, float* C,
                        int M, int N, int K, int lda, int ldb, int ldc,
                        long long aB, long long aH, long long bB, long long bH,
                        long long cB, long long cH,
                        int batch, int nh, int transB, int relu)
{
    dim3 g((N + 63) / 64, (M + 63) / 64, batch), blk(128);
    k_gemm<<<g, blk, 0, s>>>(A, B, bias, C, M, N, K, lda, ldb, ldc,
                             aB, aH, bB, bH, cB, cH, nh, transB, relu);
}

extern "C" void kernel_launch(void* const* d_in, const int* in_sizes, int n_in,
                              void* d_out, int out_size, void* d_ws, size_t ws_size,
                              hipStream_t stream)
{
    const float* trg   = (const float*)d_in[0];
    const float* src   = (const float*)d_in[1];
    const float* ft_w  = (const float*)d_in[2];
    const float* ft_b  = (const float*)d_in[3];
    const float* fc1_w = (const float*)d_in[4];
    const float* fc1_b = (const float*)d_in[5];
    const float* fc2_w = (const float*)d_in[6];
    const float* fc2_b = (const float*)d_in[7];
    const float* ln_g  = (const float*)d_in[8];
    const float* ln_b  = (const float*)d_in[9];
    const float* pf_w1 = (const float*)d_in[10];
    const float* pf_b1 = (const float*)d_in[11];
    const float* pf_w2 = (const float*)d_in[12];
    const float* pf_b2 = (const float*)d_in[13];
    const float* sa_w[4] = { (const float*)d_in[14], (const float*)d_in[16],
                             (const float*)d_in[18], (const float*)d_in[20] };
    const float* sa_b[4] = { (const float*)d_in[15], (const float*)d_in[17],
                             (const float*)d_in[19], (const float*)d_in[21] };
    const float* ea_w[4] = { (const float*)d_in[22], (const float*)d_in[24],
                             (const float*)d_in[26], (const float*)d_in[28] };
    const float* ea_b[4] = { (const float*)d_in[23], (const float*)d_in[25],
                             (const float*)d_in[27], (const float*)d_in[29] };

    const int BT = Bsz * Tt;    // 8192
    const int BS = Bsz * Ss;    // 16384
    const float INV_SCALE = 0.17677669529663687f;  // 1/sqrt(32)

    float* out    = (float*)d_out;
    float* label  = out;                  // [16,2]
    float* pooled = out + 32;             // [16,256]
    float* attP   = out + 32 + Bsz * Hh;  // [16,8,512,1024] (also score scratch)

    // workspace layout (floats)
    float* ws = (float*)d_ws;
    float* X    = ws;                         // [8192,256]
    float* T1   = X  + (size_t)BT * Hh;       // [8192,256]
    float* Qb   = T1 + (size_t)BT * Hh;       // [8192,256] (also ctx)
    float* Kb   = Qb + (size_t)BT * Hh;       // [16384,256] (aliased as FF)
    float* Vb   = Kb + (size_t)BS * Hh;       // [16384,256]
    float* wnrm = Vb + (size_t)BS * Hh;       // [16,512]
    float* f1o  = wnrm + Bsz * Tt;            // [16,256]
    float* FF   = Kb;                         // [8192,512] aliases Kb

    // x = trg @ ft_w + ft_b
    gemm(stream, trg, ft_w, ft_b, X, BT, Hh, Aa, Aa, Hh, Hh,
         0,0,0,0,0,0, 1, 1, 0, 0);

    for (int l = 0; l < 2; ++l) {
        const long long wO = (long long)l * Hh * Hh, bO = (long long)l * Hh;

        // ---------------- self-attention ----------------
        gemm(stream, X, sa_w[0] + wO, sa_b[0] + bO, Qb, BT, Hh, Hh, Hh, Hh, Hh,
             0,0,0,0,0,0, 1, 1, 0, 0);
        gemm(stream, X, sa_w[1] + wO, sa_b[1] + bO, Kb, BT, Hh, Hh, Hh, Hh, Hh,
             0,0,0,0,0,0, 1, 1, 0, 0);
        gemm(stream, X, sa_w[2] + wO, sa_b[2] + bO, Vb, BT, Hh, Hh, Hh, Hh, Hh,
             0,0,0,0,0,0, 1, 1, 0, 0);
        // scores[z] = Q_head x K_head^T  (K-dim = 32 -> 1 WMMA per fragment)
        gemm(stream, Qb, Kb, nullptr, attP, Tt, Tt, Dd, Hh, Hh, Tt,
             (long long)Tt*Hh, Dd, (long long)Tt*Hh, Dd,
             (long long)NHh*Tt*Tt, (long long)Tt*Tt,
             Bsz*NHh, NHh, 1, 0);
        k_softmax<<<Bsz*NHh*Tt, 256, 0, stream>>>(attP, Tt, INV_SCALE);
        // ctx = att x V  (into Qb)
        gemm(stream, attP, Vb, nullptr, Qb, Tt, Dd, Tt, Tt, Hh, Hh,
             (long long)NHh*Tt*Tt, (long long)Tt*Tt, (long long)Tt*Hh, Dd,
             (long long)Tt*Hh, Dd, Bsz*NHh, NHh, 0, 0);
        gemm(stream, Qb, sa_w[3] + wO, sa_b[3] + bO, T1, BT, Hh, Hh, Hh, Hh, Hh,
             0,0,0,0,0,0, 1, 1, 0, 0);
        k_add_ln<<<BT, 256, 0, stream>>>(X, T1, ln_g + bO, ln_b + bO);

        // ---------------- cross-attention ----------------
        gemm(stream, X,   ea_w[0] + wO, ea_b[0] + bO, Qb, BT, Hh, Hh, Hh, Hh, Hh,
             0,0,0,0,0,0, 1, 1, 0, 0);
        gemm(stream, src, ea_w[1] + wO, ea_b[1] + bO, Kb, BS, Hh, Hh, Hh, Hh, Hh,
             0,0,0,0,0,0, 1, 1, 0, 0);
        gemm(stream, src, ea_w[2] + wO, ea_b[2] + bO, Vb, BS, Hh, Hh, Hh, Hh, Hh,
             0,0,0,0,0,0, 1, 1, 0, 0);
        // scores -> attP in final [B,NH,T,S] layout (l=1 result IS the output)
        gemm(stream, Qb, Kb, nullptr, attP, Tt, Ss, Dd, Hh, Hh, Ss,
             (long long)Tt*Hh, Dd, (long long)Ss*Hh, Dd,
             (long long)NHh*Tt*Ss, (long long)Tt*Ss,
             Bsz*NHh, NHh, 1, 0);
        k_softmax<<<Bsz*NHh*Tt, 256, 0, stream>>>(attP, Ss, INV_SCALE);
        gemm(stream, attP, Vb, nullptr, Qb, Tt, Dd, Ss, Ss, Hh, Hh,
             (long long)NHh*Tt*Ss, (long long)Tt*Ss, (long long)Ss*Hh, Dd,
             (long long)Tt*Hh, Dd, Bsz*NHh, NHh, 0, 0);
        gemm(stream, Qb, ea_w[3] + wO, ea_b[3] + bO, T1, BT, Hh, Hh, Hh, Hh, Hh,
             0,0,0,0,0,0, 1, 1, 0, 0);
        k_add_ln<<<BT, 256, 0, stream>>>(X, T1, ln_g + bO, ln_b + bO);

        // ---------------- feed-forward ----------------
        gemm(stream, X, pf_w1 + (long long)l*Hh*PFf, pf_b1 + (long long)l*PFf,
             FF, BT, PFf, Hh, Hh, PFf, PFf, 0,0,0,0,0,0, 1, 1, 0, 1);
        gemm(stream, FF, pf_w2 + (long long)l*PFf*Hh, pf_b2 + (long long)l*Hh,
             T1, BT, Hh, PFf, PFf, Hh, Hh, 0,0,0,0,0,0, 1, 1, 0, 0);
        k_add_ln<<<BT, 256, 0, stream>>>(X, T1, ln_g + bO, ln_b + bO);
    }

    // ---------------- norm-softmax pooling + head ----------------
    k_rownorm<<<BT, 256, 0, stream>>>(X, wnrm);
    k_softmax<<<Bsz, 256, 0, stream>>>(wnrm, Tt, 1.0f);
    k_pool<<<Bsz, 256, 0, stream>>>(X, wnrm, pooled);
    gemm(stream, pooled, fc1_w, fc1_b, f1o, Bsz, Hh, Hh, Hh, Hh, Hh,
         0,0,0,0,0,0, 1, 1, 0, 1);
    k_fc2<<<1, 32, 0, stream>>>(f1o, fc2_w, fc2_b, label);
}
